// NonLocalBlock1D_7739531067712
// MI455X (gfx1250) — matrix operand order repaired and checked by reading, not compiled, not run on hardware
//
#include <hip/hip_runtime.h>
#include <hip/hip_bf16.h>

#define B_  8
#define C_  512
#define CI_ 256
#define L_  2048

typedef __bf16 bf16;
typedef __attribute__((ext_vector_type(16))) __bf16 v16bf;
typedef __attribute__((ext_vector_type(8)))  __bf16 v8bf;
typedef __attribute__((ext_vector_type(8)))  float  v8f;

union Frag16 {
    v16bf v;
    v8bf  h[2];
};

__device__ __forceinline__ v8f wmma_bf16(v16bf a, v16bf b, v8f c) {
    // D = A(16x32 bf16) * B(32x16 bf16) + C(16x16 f32)
    return __builtin_amdgcn_wmma_f32_16x16x32_bf16(
        false, a, false, b, (short)0, c, false, false);
}

// pack two floats into two bf16 lanes of one dword (v_cvt_pk_bf16_f32)
__device__ __forceinline__ unsigned pack_bf16(float a, float b) {
    union { bf16 h[2]; unsigned u; } p;
    p.h[0] = (bf16)a;
    p.h[1] = (bf16)b;
    return p.u;
}

// ---------------------------------------------------------------------------
// Kernel 0: fp32 -> bf16 weight conversion
// ---------------------------------------------------------------------------
__global__ void nlb_cvt_bf16(const float* __restrict__ s, bf16* __restrict__ d, int n) {
    int i = blockIdx.x * 256 + threadIdx.x;
    if (i < n) d[i] = (bf16)s[i];
}

// ---------------------------------------------------------------------------
// Kernel 1: fused projections  dst[b, l, d] = sum_c W[d,c] * x[b,c,l] + bias[d]
// Output bf16, row-major [B*L, CI].  blockIdx.z encodes (batch, which proj).
// Block: 256 thr = 8 waves as 4(l) x 2(d); tile 64(l) x 32(d); K = C = 512.
// x tile is transposed+converted through LDS with packed b32 stores.
// ---------------------------------------------------------------------------
__global__ __launch_bounds__(256) void nlb_proj(
    const float* __restrict__ x,
    const bf16* __restrict__ wg, const bf16* __restrict__ wth, const bf16* __restrict__ wph,
    const float* __restrict__ bg, const float* __restrict__ bth, const float* __restrict__ bph,
    bf16* __restrict__ og, bf16* __restrict__ oth, bf16* __restrict__ oph)
{
    __shared__ __align__(16) bf16 Alds[64 * 32];   // [l][c] for current 32-c slice
    unsigned* Alds32 = (unsigned*)Alds;

    const int z    = blockIdx.z;
    const int b    = z / 3;
    const int proj = z % 3;
    const bf16*  W    = (proj == 0) ? wg : (proj == 1) ? wth : wph;
    const float* bias = (proj == 0) ? bg : (proj == 1) ? bth : bph;
    bf16*        dst  = (proj == 0) ? og : (proj == 1) ? oth : oph;

    const int l0 = blockIdx.y * 64;
    const int d0 = blockIdx.x * 32;
    const int t    = threadIdx.x;
    const int lane = t & 31, wave = t >> 5;
    const int wm   = wave & 3, wn = wave >> 2;   // l-subtile, d-subtile
    const int half = lane >> 4, l16 = lane & 15;

    // staging assignment: each thread owns a (2 c) x (4 l) sub-block
    const int cp = t & 15;          // c pair index, c = 2*cp
    const int lg = (t >> 4) << 2;   // l group base, 4 rows

    v8f acc = {};
    const float* xb = x + (size_t)b * C_ * L_ + l0;

    for (int ks = 0; ks < C_ / 32; ++ks) {
        __syncthreads();
        // stage x slice [32 c][64 l] fp32 -> Alds[l][c] bf16, packed b32 stores
        {
            const float4 v0 = *(const float4*)(xb + (size_t)(ks * 32 + 2 * cp)     * L_ + lg);
            const float4 v1 = *(const float4*)(xb + (size_t)(ks * 32 + 2 * cp + 1) * L_ + lg);
            Alds32[(lg + 0) * 16 + cp] = pack_bf16(v0.x, v1.x);
            Alds32[(lg + 1) * 16 + cp] = pack_bf16(v0.y, v1.y);
            Alds32[(lg + 2) * 16 + cp] = pack_bf16(v0.z, v1.z);
            Alds32[(lg + 3) * 16 + cp] = pack_bf16(v0.w, v1.w);
        }
        __syncthreads();

        // A fragment (16x32): row m = wm*16 + l16, contiguous K chunks in LDS
        const bf16* arow = &Alds[(wm * 16 + l16) * 32];
        Frag16 a;
        a.h[0] = *(const v8bf*)(arow + 16 * half);
        a.h[1] = *(const v8bf*)(arow + 16 * half + 8);
        // B fragment (32x16): W row n = d0 + wn*16 + l16, 32 contiguous bytes
        const bf16* brow = W + (size_t)(d0 + wn * 16 + l16) * C_ + ks * 32 + 16 * half;
        Frag16 bb;
        bb.h[0] = *(const v8bf*)brow;
        bb.h[1] = *(const v8bf*)(brow + 8);

        acc = wmma_bf16(a.v, bb.v, acc);
    }

    const int   n  = d0 + wn * 16 + l16;
    const float bv = bias[n];
    const int   lr = l0 + wm * 16;
    #pragma unroll
    for (int r = 0; r < 8; ++r) {
        int m = r + 8 * half;
        dst[(size_t)(b * L_ + lr + m) * CI_ + n] = (bf16)(acc[r] + bv);
    }
}

// ---------------------------------------------------------------------------
// Kernel 2: flash attention.  Q=theta, K=phi, V=g, all bf16 [B*L, CI].
// Block 256 thr = 8 waves: 4 q-subtiles (16 rows) x 2 d-halves (128 cols).
// Key tile = 64 (4 S-subtiles): 48 WMMAs per online-softmax pass.
// f (L x L) never touches HBM.
// ---------------------------------------------------------------------------
__global__ __launch_bounds__(256) void nlb_attn(
    const bf16* __restrict__ Q,
    const bf16* __restrict__ K,
    const bf16* __restrict__ V,
    bf16* __restrict__ Y)
{
    __shared__ __align__(16) bf16 Vlds[CI_ * 64];   // [d][k]  32 KB
    __shared__ __align__(16) bf16 Plds[8][16 * 64]; // per-wave P tile, 16 KB
    unsigned* Vlds32 = (unsigned*)Vlds;

    const int b  = blockIdx.y;
    const int q0 = blockIdx.x * 64;
    const int t    = threadIdx.x;
    const int lane = t & 31, wave = t >> 5;
    const int wq   = wave & 3, wd = wave >> 2;      // q-subtile, d-half
    const int half = lane >> 4, l16 = lane & 15;

    // preload this wave's Q fragments (16 rows x 256 K) : 8 fragments
    const bf16* Qb = Q + (size_t)(b * L_ + q0 + wq * 16 + l16) * CI_;
    Frag16 qf[8];
    #pragma unroll
    for (int ks = 0; ks < 8; ++ks) {
        qf[ks].h[0] = *(const v8bf*)(Qb + ks * 32 + 16 * half);
        qf[ks].h[1] = *(const v8bf*)(Qb + ks * 32 + 16 * half + 8);
    }

    v8f zero = {};
    v8f acc[8];
    #pragma unroll
    for (int i = 0; i < 8; ++i) acc[i] = zero;
    float mrow[8], lrow[8];
    #pragma unroll
    for (int r = 0; r < 8; ++r) { mrow[r] = -1e30f; lrow[r] = 0.0f; }

    const bf16* Kb = K + (size_t)b * L_ * CI_;
    const bf16* Vb = V + (size_t)b * L_ * CI_;

    for (int kt = 0; kt < L_ / 64; ++kt) {
        const int key0 = kt * 64;

        // speculative prefetch of next tile's K/V rows (global_prefetch_b8)
        if (kt + 1 < L_ / 64) {
            __builtin_prefetch(Kb + (size_t)(key0 + 64 + l16) * CI_, 0, 1);
            __builtin_prefetch(Vb + (size_t)(key0 + 64 + l16) * CI_, 0, 1);
        }

        __syncthreads();
        // stage V tile [64 key][256 d] -> Vlds[d][k] transposed, packed b32 stores
        #pragma unroll
        for (int rep = 0; rep < 4; ++rep) {
            int idx = t + rep * 256;             // 0..1023
            int kp  = idx & 31;                  // key pair, k = 2*kp
            int dg  = (idx >> 5) << 3;           // d group of 8
            v8bf va = *(const v8bf*)(Vb + (size_t)(key0 + 2 * kp)     * CI_ + dg);
            v8bf vb = *(const v8bf*)(Vb + (size_t)(key0 + 2 * kp + 1) * CI_ + dg);
            #pragma unroll
            for (int j = 0; j < 8; ++j) {
                union { bf16 h[2]; unsigned u; } p;
                p.h[0] = va[j];
                p.h[1] = vb[j];
                Vlds32[(dg + j) * 32 + kp] = p.u;
            }
        }
        __syncthreads();

        // S = Q * K^T : four 16x16 key-subtiles, contraction over CI
        v8f s[4];
        #pragma unroll
        for (int j = 0; j < 4; ++j) s[j] = zero;
        #pragma unroll
        for (int ks = 0; ks < 8; ++ks) {
            #pragma unroll
            for (int j = 0; j < 4; ++j) {
                const bf16* kr = Kb + (size_t)(key0 + j * 16 + l16) * CI_ + ks * 32 + 16 * half;
                Frag16 kb;
                kb.h[0] = *(const v8bf*)kr;
                kb.h[1] = *(const v8bf*)(kr + 8);
                s[j] = wmma_bf16(qf[ks].v, kb.v, s[j]);
            }
        }

        // online softmax over the 64 keys; row m = r + 8*half lives in one
        // 16-lane group, so xor-shuffles with mask<16 reduce within the row.
        bf16* Pw = &Plds[wave][0];
        float alpha[8];
        #pragma unroll
        for (int r = 0; r < 8; ++r) {
            float v0 = s[0][r], v1 = s[1][r], v2 = s[2][r], v3 = s[3][r];
            float mx = fmaxf(fmaxf(v0, v1), fmaxf(v2, v3));
            #pragma unroll
            for (int off = 1; off < 16; off <<= 1)
                mx = fmaxf(mx, __shfl_xor(mx, off, 32));
            float mnew = fmaxf(mrow[r], mx);
            float al   = __expf(mrow[r] - mnew);
            float p0 = __expf(v0 - mnew);
            float p1 = __expf(v1 - mnew);
            float p2 = __expf(v2 - mnew);
            float p3 = __expf(v3 - mnew);
            float rs = (p0 + p1) + (p2 + p3);
            #pragma unroll
            for (int off = 1; off < 16; off <<= 1)
                rs += __shfl_xor(rs, off, 32);
            lrow[r]  = lrow[r] * al + rs;
            mrow[r]  = mnew;
            alpha[r] = al;
            int m = r + 8 * half;
            Pw[m * 64 + l16]      = (bf16)p0;
            Pw[m * 64 + 16 + l16] = (bf16)p1;
            Pw[m * 64 + 32 + l16] = (bf16)p2;
            Pw[m * 64 + 48 + l16] = (bf16)p3;
        }
        #pragma unroll
        for (int i = 0; i < 8; ++i) {
            #pragma unroll
            for (int r = 0; r < 8; ++r) acc[i][r] *= alpha[r];
        }

        // wave-private LDS round trip: wait our DS stores, then read P as A-frags
        asm volatile("s_wait_dscnt 0x0" ::: "memory");
        Frag16 pf0, pf1;
        const bf16* prow = Pw + l16 * 64;
        pf0.h[0] = *(const v8bf*)(prow + 16 * half);
        pf0.h[1] = *(const v8bf*)(prow + 16 * half + 8);
        pf1.h[0] = *(const v8bf*)(prow + 32 + 16 * half);
        pf1.h[1] = *(const v8bf*)(prow + 32 + 16 * half + 8);

        // O += P(16x64) * V(64x128-half) : two k-steps per d-subtile
        #pragma unroll
        for (int i = 0; i < 8; ++i) {
            int d = wd * 128 + i * 16 + l16;
            const bf16* vrow = &Vlds[d * 64];
            Frag16 vf0, vf1;
            vf0.h[0] = *(const v8bf*)(vrow + 16 * half);
            vf0.h[1] = *(const v8bf*)(vrow + 16 * half + 8);
            vf1.h[0] = *(const v8bf*)(vrow + 32 + 16 * half);
            vf1.h[1] = *(const v8bf*)(vrow + 32 + 16 * half + 8);
            acc[i] = wmma_bf16(pf0.v, vf0.v, acc[i]);
            acc[i] = wmma_bf16(pf1.v, vf1.v, acc[i]);
        }
    }

    // epilogue: O /= l ; store bf16 y[B*L, CI]
    #pragma unroll
    for (int i = 0; i < 8; ++i) {
        int d = wd * 128 + i * 16 + l16;
        #pragma unroll
        for (int r = 0; r < 8; ++r) {
            int m = r + 8 * half;
            Y[(size_t)(b * L_ + q0 + wq * 16 + m) * CI_ + d] = (bf16)(acc[i][r] / lrow[r]);
        }
    }
}

// ---------------------------------------------------------------------------
// Kernel 3: out[b,o,l] = sum_d Wo[o,d] * y[b,l,d] + bo[o] + x[b,o,l]
// Both operands contiguous per lane -> no LDS needed. Tile 64(o) x 32(l).
// ---------------------------------------------------------------------------
__global__ __launch_bounds__(256) void nlb_out(
    const bf16* __restrict__ Wo,
    const float* __restrict__ bo,
    const bf16* __restrict__ Yb,
    const float* __restrict__ x,
    float* __restrict__ out)
{
    const int b  = blockIdx.z;
    const int o0 = blockIdx.y * 64;
    const int l0 = blockIdx.x * 32;
    const int t    = threadIdx.x;
    const int lane = t & 31, wave = t >> 5;
    const int wo   = wave & 3, wl = wave >> 2;
    const int half = lane >> 4, l16 = lane & 15;

    v8f acc = {};
    const bf16* wrow = Wo + (size_t)(o0 + wo * 16 + l16) * CI_;
    const bf16* yrow = Yb + (size_t)(b * L_ + l0 + wl * 16 + l16) * CI_;
    #pragma unroll
    for (int ks = 0; ks < 8; ++ks) {
        Frag16 a, bb;
        a.h[0]  = *(const v8bf*)(wrow + ks * 32 + 16 * half);
        a.h[1]  = *(const v8bf*)(wrow + ks * 32 + 16 * half + 8);
        bb.h[0] = *(const v8bf*)(yrow + ks * 32 + 16 * half);
        bb.h[1] = *(const v8bf*)(yrow + ks * 32 + 16 * half + 8);
        acc = wmma_bf16(a.v, bb.v, acc);
    }

    const int l = l0 + wl * 16 + l16;
    #pragma unroll
    for (int r = 0; r < 8; ++r) {
        int o = o0 + wo * 16 + r + 8 * half;
        size_t idx = ((size_t)b * C_ + o) * L_ + l;
        out[idx] = acc[r] + bo[o] + x[idx];
    }
}

// ---------------------------------------------------------------------------
extern "C" void kernel_launch(void* const* d_in, const int* in_sizes, int n_in,
                              void* d_out, int out_size, void* d_ws, size_t ws_size,
                              hipStream_t stream) {
    const float* x    = (const float*)d_in[0];
    const float* g_w  = (const float*)d_in[1];
    const float* g_b  = (const float*)d_in[2];
    const float* th_w = (const float*)d_in[3];
    const float* th_b = (const float*)d_in[4];
    const float* ph_w = (const float*)d_in[5];
    const float* ph_b = (const float*)d_in[6];
    const float* o_w  = (const float*)d_in[7];
    const float* o_b  = (const float*)d_in[8];
    float* out = (float*)d_out;

    char* ws = (char*)d_ws;
    const size_t WSZ  = (size_t)CI_ * C_ * sizeof(bf16);      // 256 KB per weight
    const size_t PSZ  = (size_t)B_ * L_ * CI_ * sizeof(bf16); // 8 MB per projection
    bf16* wg_bf  = (bf16*)(ws);
    bf16* wth_bf = (bf16*)(ws + WSZ);
    bf16* wph_bf = (bf16*)(ws + 2 * WSZ);
    bf16* wo_bf  = (bf16*)(ws + 3 * WSZ);
    bf16* theta  = (bf16*)(ws + 4 * WSZ);
    bf16* phi    = (bf16*)(ws + 4 * WSZ + PSZ);
    bf16* gmat   = (bf16*)(ws + 4 * WSZ + 2 * PSZ);
    bf16* ymat   = (bf16*)(ws + 4 * WSZ + 3 * PSZ);

    const int nw = CI_ * C_;   // all four weight matrices are 256*512
    nlb_cvt_bf16<<<(nw + 255) / 256, 256, 0, stream>>>(g_w,  wg_bf,  nw);
    nlb_cvt_bf16<<<(nw + 255) / 256, 256, 0, stream>>>(th_w, wth_bf, nw);
    nlb_cvt_bf16<<<(nw + 255) / 256, 256, 0, stream>>>(ph_w, wph_bf, nw);
    nlb_cvt_bf16<<<(nw + 255) / 256, 256, 0, stream>>>(o_w,  wo_bf,  nw);

    // projections: grid (CI/32, L/64, B*3)
    nlb_proj<<<dim3(CI_ / 32, L_ / 64, B_ * 3), 256, 0, stream>>>(
        x, wg_bf, wth_bf, wph_bf, g_b, th_b, ph_b, gmat, theta, phi);

    // flash attention: grid (L/64, B)
    nlb_attn<<<dim3(L_ / 64, B_), 256, 0, stream>>>(theta, phi, gmat, ymat);

    // output projection + residual: grid (L/32, C/64, B)
    nlb_out<<<dim3(L_ / 32, C_ / 64, B_), 256, 0, stream>>>(
        wo_bf, o_b, ymat, x, out);

    (void)in_sizes; (void)n_in; (void)out_size; (void)ws_size;
}
